// PeriodicityTransform_25237227831544
// MI455X (gfx1250) — matrix operand order repaired and checked by reading, not compiled, not run on hardware
//
#include <hip/hip_runtime.h>
#include <hip/hip_bf16.h>

// Problem constants from the reference setup_inputs(): x = [32, 1024, 128] f32, k = 5.
#define BB   32
#define TQ   1024
#define NCH  128
#define KK   5
#define FQ   512          // usable frequency bins f = 1..512 (DC zeroed, bin 512 = Nyquist)
#define NSER (BB * NCH)   // 4096 independent time series
#define RPB  32           // series (rows) per workgroup: 2 WMMA row-tiles
#define APAD 8            // LDS row padding (halves): stride 2064 B == 4 banks/row

typedef __attribute__((ext_vector_type(16))) _Float16 v16h;
typedef __attribute__((ext_vector_type(8)))  float    v8f;

// ---------------------------------------------------------------------------
// Kernel 1: transpose x[b, t, n] (f32) -> Xt[b*NCH + n][t] (f32, exact fold
// arithmetic) and Xh (f16, WMMA A operand), LDS-tiled for coalescing.
// ---------------------------------------------------------------------------
__global__ __launch_bounds__(1024) void transpose_kernel(
    const float* __restrict__ x, float* __restrict__ Xt,
    _Float16* __restrict__ Xh) {
  __shared__ float tile[32][33];
  const int t0 = blockIdx.x * 32;
  const int n0 = blockIdx.y * 32;
  const int b  = blockIdx.z;
  const int tx = threadIdx.x, ty = threadIdx.y;
  tile[ty][tx] = x[((size_t)b * TQ + (t0 + ty)) * NCH + (n0 + tx)];
  __syncthreads();
  const float v = tile[tx][ty];
  const size_t o = ((size_t)b * NCH + (n0 + ty)) * TQ + (t0 + tx);
  Xt[o] = v;
  Xh[o] = (_Float16)v;
}

// ---------------------------------------------------------------------------
// Kernel 2: DFT twiddle matrix, f16, COLUMN-major: Wt[c][t], c = 0..1023.
// Column c encodes frequency f = c/2 + 1; even c = cos term, odd c = sin term
// (sign of the sin term is irrelevant for |X|^2). Integer phase reduction mod
// T keeps f32 trig accurate. Column-major => each WMMA B-fragment lane load
// is one contiguous 32-byte chunk.
// ---------------------------------------------------------------------------
__global__ __launch_bounds__(256) void twiddle_kernel(_Float16* __restrict__ Wt) {
  const int c = blockIdx.x;            // 0..1023
  const int f = (c >> 1) + 1;          // 1..512
  const bool isSin = (c & 1);
  for (int t = threadIdx.x; t < TQ; t += 256) {
    int ph = (f * t) & (TQ - 1);       // (f*t) mod 1024, exact in int
    float ang = (float)ph * (6.28318530717958647692f / (float)TQ);
    float sv, cv;
    __sincosf(ang, &sv, &cv);
    Wt[(size_t)c * TQ + t] = (_Float16)(isSin ? sv : cv);
  }
}

// ---------------------------------------------------------------------------
// Kernel 3 (fused), one 8-wave WG per 32 series:
//   (a) async-preload 32x1024 f16 samples into LDS (ASYNCcnt engine),
//   (b) power spectrum via WMMA GEMM: 2 row-tiles share every B fragment,
//   (c) top-5 bins per series -> (P, start, 1/cycles),
//   (d) fold tail into P buckets (ds_add_f32), write mean to out.
// Phase-1 LDS (Ash) is dead after the K-loop, so phase-2 data overlays it.
// ---------------------------------------------------------------------------
union SharedU {
  _Float16 Ash[RPB][TQ + APAD];                  // 66048 B, GEMM phase
  struct {
    float mag[RPB][FQ];                          // 65536 B
    float buck[TQ];                              // 4096 B
    int   P[RPB][KK];
    int   start[RPB][KK];
    float invc[RPB][KK];
  } p2;                                          // ~71.5 KB, analysis phase
};

__global__ __launch_bounds__(256) void spectral_fold_kernel(
    const float* __restrict__ Xt, const _Float16* __restrict__ Xh,
    const _Float16* __restrict__ Wt, float* __restrict__ out, int Pmax) {
  __shared__ SharedU u;

  const int tid  = threadIdx.x;
  const int lane = tid & 31;
  const int w    = tid >> 5;             // wave id 0..7
  const int lm   = lane & 15;            // A-row / B&D-column within tile
  const int hi   = lane >> 4;            // lane half
  const int rowBase = blockIdx.x * RPB;  // first series of this block

  // ---- (a) one-shot async preload: 32 rows x 2 KB -> LDS (128 16B chunks/row)
  {
    const char* gbase = (const char*)(Xh + (size_t)rowBase * TQ);
    #pragma unroll
    for (int it = 0; it < (RPB * 128) / 256; ++it) {
      const int g  = tid + it * 256;     // chunk id 0..4095
      const int r  = g >> 7;             // row 0..31
      const int co = g & 127;            // 16B chunk within row
      unsigned lds_off = (unsigned)(uintptr_t)&u.Ash[r][co * 8];
      const char* gp = gbase + ((size_t)r * TQ + (size_t)co * 8) * sizeof(_Float16);
      asm volatile("global_load_async_to_lds_b128 %0, %1, off"
                   :: "v"(lds_off), "v"(gp) : "memory");
    }
    asm volatile("s_wait_asynccnt 0x0" ::: "memory");
  }
  __syncthreads();

  // ---- (b) K-loop: no barriers, 16 wmma per 8 B-fragment loads
  v8f acc0[8], acc1[8];
  const v8f vzero = {};
  #pragma unroll
  for (int i = 0; i < 8; ++i) { acc0[i] = vzero; acc1[i] = vzero; }

  for (int kb = 0; kb < TQ; kb += 32) {
    // A fragments, contiguous K order (matches B):
    // lanes 0-15: elems 0-15 -> K 0-15
    // lanes 16-31: elems 0-15 -> K 16-31
    v16h a0, a1;
    {
      const _Float16* ar0 = &u.Ash[lm][kb];
      const _Float16* ar1 = &u.Ash[16 + lm][kb];
      const int k0 = hi * 16;
      #pragma unroll
      for (int e = 0; e < 16; ++e) {
        a0[e]     = ar0[k0 + e];
        a1[e]     = ar1[k0 + e];
      }
    }
    #pragma unroll
    for (int i = 0; i < 8; ++i) {
      // B fragment: lane column N = lm, elems e -> K = 16*hi + e: one
      // contiguous 32 B chunk of the column-major twiddle matrix.
      const int col = (w * 8 + i) * 16 + lm;
      const _Float16* bp = Wt + (size_t)col * TQ + kb + hi * 16;
      __builtin_prefetch((const char*)bp + 64, 0, 3);  // next K chunk
      v16h bfrag = *(const v16h*)bp;
      acc0[i] = __builtin_amdgcn_wmma_f32_16x16x32_f16(
          false, a0, false, bfrag, (short)0, acc0[i], false, false);
      acc1[i] = __builtin_amdgcn_wmma_f32_16x16x32_f16(
          false, a1, false, bfrag, (short)0, acc1[i], false, false);
    }
  }
  __syncthreads();   // Ash dead everywhere before mag overlays it

  // ---- |X|^2: D layout has col N = lm per lane, row M = r + 8*hi in VGPR r.
  // cos (even col) and sin (odd col) sit in adjacent lanes -> shfl_xor(1).
  #pragma unroll
  for (int i = 0; i < 8; ++i) {
    #pragma unroll
    for (int r = 0; r < 8; ++r) {
      float v0 = acc0[i][r], v1 = acc1[i][r];
      float m0 = v0 * v0;  m0 += __shfl_xor(m0, 1, 32);
      float m1 = v1 * v1;  m1 += __shfl_xor(m1, 1, 32);
      if ((lane & 1) == 0) {
        const int fi = w * 64 + i * 8 + (lm >> 1);   // 0..511 (bin f = fi+1)
        u.p2.mag[r + 8 * hi][fi]       = m0;
        u.p2.mag[16 + r + 8 * hi][fi]  = m1;
      }
    }
  }
  __syncthreads();

  // ---- (c) top-5 per series (strict '>' => stable, lower index wins ties,
  // matching lax.top_k; bins 1..512 are exactly fi 0..511 here, DC excluded).
  if (tid < RPB) {
    float bv[KK]; int bidx[KK];
    #pragma unroll
    for (int j = 0; j < KK; ++j) { bv[j] = -1.0f; bidx[j] = 0; }
    for (int fi = 0; fi < FQ; ++fi) {
      float m = u.p2.mag[tid][fi];
      if (m > bv[KK - 1]) {
        int j = KK - 1;
        while (j > 0 && m > bv[j - 1]) {
          bv[j] = bv[j - 1]; bidx[j] = bidx[j - 1]; --j;
        }
        bv[j] = m; bidx[j] = fi;
      }
    }
    #pragma unroll
    for (int j = 0; j < KK; ++j) {
      int kidx = bidx[j] + 1;                 // >= 1 automatically
      int P = TQ / kidx;  if (P < 1) P = 1;   // 2..1024
      int cyc = TQ / P;   if (cyc < 1) cyc = 1;
      int take = cyc * P;
      u.p2.P[tid][j]     = P;
      u.p2.start[tid][j] = TQ - take;
      u.p2.invc[tid][j]  = 1.0f / (float)cyc;
    }
  }
  __syncthreads();

  // ---- (d) fold: bucket-sum the tail in LDS (ds_add_f32), write mean.
  // out layout [B, KK, Pmax, N]; positions p >= P stay 0 from the memset.
  for (int task = 0; task < RPB * KK; ++task) {
    const int r = task / KK, j = task % KK;
    const int P = u.p2.P[r][j], start = u.p2.start[r][j];
    const float invc = u.p2.invc[r][j];
    for (int p = tid; p < P; p += 256) u.p2.buck[p] = 0.0f;
    __syncthreads();
    const float* xs = Xt + (size_t)(rowBase + r) * TQ;
    for (int t = start + tid; t < TQ; t += 256) {
      int p = (t - start) % P;
      atomicAdd(&u.p2.buck[p], xs[t]);       // ds_add_f32
    }
    __syncthreads();
    const int s = rowBase + r;
    const int b = s >> 7, n = s & (NCH - 1);
    float* op = out + ((size_t)(b * KK + j) * (size_t)Pmax) * NCH + n;
    for (int p = tid; p < P; p += 256) op[(size_t)p * NCH] = u.p2.buck[p] * invc;
    __syncthreads();
  }
}

// ---------------------------------------------------------------------------
// Host launcher. Workspace: Xt f32 (16 MiB) | Xh f16 (8 MiB) | Wt f16 (2 MiB).
// Pmax is recovered from out_size (out = [B, KK, Pmax, N] f32).
// ---------------------------------------------------------------------------
extern "C" void kernel_launch(void* const* d_in, const int* in_sizes, int n_in,
                              void* d_out, int out_size, void* d_ws, size_t ws_size,
                              hipStream_t stream) {
  const float* x = (const float*)d_in[0];      // [B, T, N] f32; d_in[1] = k (==5)
  float* out = (float*)d_out;
  const int Pmax = out_size / (BB * KK * NCH);

  float*    Xt = (float*)d_ws;
  _Float16* Xh = (_Float16*)((char*)d_ws + (size_t)NSER * TQ * sizeof(float));
  _Float16* Wt = Xh + (size_t)NSER * TQ;

  // Zero output: implements zero-padding of bucket positions p >= P.
  hipMemsetAsync(d_out, 0, (size_t)out_size * sizeof(float), stream);

  transpose_kernel<<<dim3(TQ / 32, NCH / 32, BB), dim3(32, 32), 0, stream>>>(x, Xt, Xh);
  twiddle_kernel<<<dim3(2 * FQ), dim3(256), 0, stream>>>(Wt);
  spectral_fold_kernel<<<dim3(NSER / RPB), dim3(256), 0, stream>>>(Xt, Xh, Wt, out, Pmax);
}
